// spk_vq_vae_resnet_45561013076383
// MI455X (gfx1250) — compile-verified, matrix operand order checked by reading, not run on hardware
//
#include <hip/hip_runtime.h>
#include <hip/hip_bf16.h>

#define NGRP 8
#define GN_EPS 1e-5f

typedef __attribute__((ext_vector_type(16))) __bf16 v16bf;
typedef __attribute__((ext_vector_type(8)))  __bf16 v8bf;
typedef __attribute__((ext_vector_type(8)))  float  v8f;

__device__ __forceinline__ float selu_f(float x) {
  const float a = 1.6732632423543772f, s = 1.0507009873554805f;
  return x > 0.f ? s * x : s * a * (expf(x) - 1.f);
}

__device__ __forceinline__ v16bf frag_combine(v8bf lo, v8bf hi) {
  v16bf r;
#pragma unroll
  for (int i = 0; i < 8; ++i) { r[i] = lo[i]; r[i + 8] = hi[i]; }
  return r;
}

// ---------------- direct fp32 conv (tiny channel counts / grouped) ----------------
__global__ void k_conv_direct(const float* __restrict__ in, const float* __restrict__ w,
                              const float* __restrict__ bias, float* __restrict__ out,
                              int B, int Cin, int Cout, int T, int K, int groups) {
  int idx = blockIdx.x * blockDim.x + threadIdx.x;
  int total = B * Cout * T;
  if (idx >= total) return;
  int t = idx % T; int rc = idx / T; int co = rc % Cout; int b = rc / Cout;
  int cig = Cin / groups, cog = Cout / groups;
  int g = co / cog, p = (K - 1) >> 1;
  float acc = bias ? bias[co] : 0.f;
  for (int ci = 0; ci < cig; ++ci) {
    const float* ip = in + (b * Cin + g * cig + ci) * T;
    const float* wp = w + (co * cig + ci) * K;
    for (int k = 0; k < K; ++k) {
      int ts = t + k - p;
      if (ts >= 0 && ts < T) acc = fmaf(ip[ts], wp[k], acc);
    }
  }
  out[idx] = acc;
}

// ---------------- implicit-GEMM conv via bf16 WMMA ----------------
// xT : bf16 transposed activations, xT[(b*T + t)*Cin + ci]
// wb : bf16 frag-packed weights,    wb[(co*K + k)*Cin + ci]
// out: f32 (b, co, t); optional affine (scale,shift per co); optional bf16-T out.
// One wave computes a 16(co) x 64(t) tile: A fragment reused across 4 B fragments.
// grid (T/64, Cout/16, B), block 32
__global__ __launch_bounds__(32)
void k_conv_wmma(const __bf16* __restrict__ xT, const __bf16* __restrict__ wb,
                 const float* __restrict__ scale, const float* __restrict__ shift,
                 float* __restrict__ out, __bf16* __restrict__ outT,
                 int Cin, int Cout, int T, int K) {
  int t0 = blockIdx.x * 64, co0 = blockIdx.y * 16, b = blockIdx.z;
  int lane = threadIdx.x;
  int nlo = lane & 15;
  int kbase = (lane < 16) ? 0 : 8;   // ISA 16-bit fragment K-subset base
  int p = (K - 1) >> 1;
  int m = co0 + nlo;                 // A-frag row (output channel)
  v8f acc[4];
#pragma unroll
  for (int j = 0; j < 4; ++j) acc[j] = (v8f){0.f,0.f,0.f,0.f,0.f,0.f,0.f,0.f};

  for (int k = 0; k < K; ++k) {
    const __bf16* wrow = wb + (m * K + k) * Cin;
    for (int c0 = 0; c0 < Cin; c0 += 32) {
      if (c0 + 32 < Cin) __builtin_prefetch(wrow + c0 + 32, 0, 0);
      v8bf alo = *(const v8bf*)(wrow + c0 + kbase);
      v8bf ahi = *(const v8bf*)(wrow + c0 + 16 + kbase);
      v16bf af = frag_combine(alo, ahi);
#pragma unroll
      for (int j = 0; j < 4; ++j) {
        int n = t0 + j * 16 + nlo;
        int tsrc = n + k - p;
        v16bf bfr;
        if (tsrc >= 0 && tsrc < T) {
          const __bf16* xrow = xT + (tsrc + b * T) * Cin + c0;
          v8bf blo = *(const v8bf*)(xrow + kbase);
          v8bf bhi = *(const v8bf*)(xrow + 16 + kbase);
          bfr = frag_combine(blo, bhi);
        } else {
#pragma unroll
          for (int i = 0; i < 16; ++i) bfr[i] = (__bf16)0.f;
        }
        acc[j] = __builtin_amdgcn_wmma_f32_16x16x32_bf16(false, af, false, bfr,
                                                         (short)0, acc[j], false, false);
      }
    }
  }
  int mb = co0 + ((lane < 16) ? 0 : 8);
#pragma unroll
  for (int j = 0; j < 4; ++j) {
    int n = t0 + j * 16 + nlo;
    float vals[8];
#pragma unroll
    for (int i = 0; i < 8; ++i) {
      float v = acc[j][i];
      if (scale) v = fmaf(v, scale[mb + i], shift[mb + i]);
      vals[i] = v;
      out[(b * Cout + mb + i) * T + n] = v;
    }
    if (outT) {
      v8bf ot;
#pragma unroll
      for (int i = 0; i < 8; ++i) ot[i] = (__bf16)vals[i];
      *(v8bf*)(outT + (b * T + n) * Cout + mb) = ot;
    }
  }
}

// ---------------- GroupNorm (+residual) (+SELU), optional bf16-T copy ----------------
// grid (NGRP, B), block 256
__global__ __launch_bounds__(256)
void k_gn(const float* __restrict__ x, const float* __restrict__ res,
          float* __restrict__ out, __bf16* __restrict__ outT,
          const float* __restrict__ gamma, const float* __restrict__ beta,
          int C, int T, int do_selu) {
  int g = blockIdx.x, b = blockIdx.y;
  int cg = C / NGRP;
  int n = cg * T;
  int off = (b * C + g * cg) * T;
  const float* xp = x + off;
  const float* rp = res ? res + off : nullptr;
  float s = 0.f, ss = 0.f;
  for (int i = threadIdx.x; i < n; i += blockDim.x) { float v = xp[i]; s += v; ss += v * v; }
  __shared__ float sh_s[256], sh_ss[256];
  sh_s[threadIdx.x] = s; sh_ss[threadIdx.x] = ss;
  __syncthreads();
  for (int st = 128; st > 0; st >>= 1) {
    if ((int)threadIdx.x < st) {
      sh_s[threadIdx.x]  += sh_s[threadIdx.x + st];
      sh_ss[threadIdx.x] += sh_ss[threadIdx.x + st];
    }
    __syncthreads();
  }
  float mean = sh_s[0] / (float)n;
  float var  = sh_ss[0] / (float)n - mean * mean;
  float rstd = rsqrtf(var + GN_EPS);
  for (int i = threadIdx.x; i < n; i += blockDim.x) {
    int c = g * cg + i / T;
    int t = i % T;
    float v = (xp[i] - mean) * rstd * gamma[c] + beta[c];
    if (rp) v += rp[i];
    if (do_selu) v = selu_f(v);
    out[off + i] = v;
    if (outT) outT[(b * T + t) * C + c] = (__bf16)v;
  }
}

// ---------------- pooling / upsample (f32 out + optional bf16-T out) ----------------
__global__ void k_pool(const float* __restrict__ in, float* __restrict__ out,
                       __bf16* __restrict__ outT, int total, int Tout, int C) {
  int i = blockIdx.x * blockDim.x + threadIdx.x;
  if (i >= total) return;
  int t = i % Tout, bc = i / Tout;
  const float* p = in + bc * (2 * Tout) + 2 * t;
  float v = fmaxf(p[0], p[1]);
  out[i] = v;
  if (outT) {
    int c = bc % C, b = bc / C;
    outT[(b * Tout + t) * C + c] = (__bf16)v;
  }
}
__global__ void k_up(const float* __restrict__ in, float* __restrict__ out,
                     __bf16* __restrict__ outT, int total, int Tout, int C) {
  int i = blockIdx.x * blockDim.x + threadIdx.x;
  if (i >= total) return;
  int t = i % Tout, bc = i / Tout;
  float v = in[bc * (Tout >> 1) + (t >> 1)];
  out[i] = v;
  if (outT) {
    int c = bc % C, b = bc / C;
    outT[(b * Tout + t) * C + c] = (__bf16)v;
  }
}

// ---------------- VQ helpers ----------------
__global__ void k_eprep(const float* __restrict__ e, __bf16* __restrict__ eb) {
  int i = blockIdx.x * blockDim.x + threadIdx.x;
  if (i < 1024 * 64) eb[i] = (__bf16)e[i];
}
__global__ void k_ww(const float* __restrict__ e, float* __restrict__ ww) {
  int n = blockIdx.x * blockDim.x + threadIdx.x;
  if (n >= 1024) return;
  float s = 0.f;
  for (int d = 0; d < 64; ++d) { float v = (float)(__bf16)e[n * 64 + d]; s += v * v; }
  ww[n] = s;
}
__global__ void k_zero(float* p, int nv) {
  int i = blockIdx.x * blockDim.x + threadIdx.x;
  if (i < nv) p[i] = 0.f;
}

// fused distance GEMM + argmin; one wave per 16 Z-rows.
// zT: bf16 (b,t,c) rows of Z; eb: bf16 codebook (1024,64)
__global__ __launch_bounds__(32)
void k_vq(const __bf16* __restrict__ zT, const __bf16* __restrict__ eb,
          const float* __restrict__ ww, int T, int* __restrict__ jout,
          float* __restrict__ lacc) {
  int lane = threadIdx.x;
  int nlo = lane & 15;
  int kbase = (lane < 16) ? 0 : 8;
  int r0 = blockIdx.x * 16;
  int r = r0 + nlo, b = r / T, t = r % T;
  const __bf16* zp = zT + (b * T + t) * 64;
  v16bf a0 = frag_combine(*(const v8bf*)(zp + kbase),      *(const v8bf*)(zp + 16 + kbase));
  v16bf a1 = frag_combine(*(const v8bf*)(zp + 32 + kbase), *(const v8bf*)(zp + 48 + kbase));
  float ssq = 0.f;
#pragma unroll
  for (int i = 0; i < 16; ++i) {
    float f0 = (float)a0[i], f1 = (float)a1[i];
    ssq += f0 * f0 + f1 * f1;
  }
  ssq += __shfl_xor(ssq, 16, 32);
  __shared__ float zz_s[16];
  if (lane < 16) zz_s[lane] = ssq;
  __syncthreads();
  float bd[8]; int bj[8];
#pragma unroll
  for (int i = 0; i < 8; ++i) { bd[i] = 3.4e38f; bj[i] = 0; }
  for (int n0 = 0; n0 < 1024; n0 += 16) {
    int code = n0 + nlo;
    const __bf16* ep = eb + code * 64;
    v16bf b0 = frag_combine(*(const v8bf*)(ep + kbase),      *(const v8bf*)(ep + 16 + kbase));
    v16bf b1 = frag_combine(*(const v8bf*)(ep + 32 + kbase), *(const v8bf*)(ep + 48 + kbase));
    v8f acc = {0.f, 0.f, 0.f, 0.f, 0.f, 0.f, 0.f, 0.f};
    acc = __builtin_amdgcn_wmma_f32_16x16x32_bf16(false, a0, false, b0, (short)0, acc, false, false);
    acc = __builtin_amdgcn_wmma_f32_16x16x32_bf16(false, a1, false, b1, (short)0, acc, false, false);
    float wwv = ww[code];
#pragma unroll
    for (int i = 0; i < 8; ++i) {
      float d = zz_s[kbase + i] - 2.f * acc[i] + wwv;
      if (d < bd[i]) { bd[i] = d; bj[i] = code; }
    }
  }
#pragma unroll
  for (int mask = 1; mask <= 8; mask <<= 1) {
#pragma unroll
    for (int i = 0; i < 8; ++i) {
      float od = __shfl_xor(bd[i], mask, 32);
      int   oj = __shfl_xor(bj[i], mask, 32);
      if (od < bd[i] || (od == bd[i] && oj < bj[i])) { bd[i] = od; bj[i] = oj; }
    }
  }
  if (nlo == 0) {
    float s = 0.f;
#pragma unroll
    for (int i = 0; i < 8; ++i) { jout[r0 + kbase + i] = bj[i]; s += bd[i]; }
    atomicAdd(lacc, s);
  }
}

// gather codebook rows -> bf16-T hq: hqT[(b*T+t)*64 + c]
__global__ void k_gather(const int* __restrict__ j, const float* __restrict__ e,
                         __bf16* __restrict__ hqT, int T) {
  int i = blockIdx.x * blockDim.x + threadIdx.x;
  int total = 128 * 64 * T;
  if (i >= total) return;
  int c = i % 64; int rt = i / 64; int t = rt % T; int b = rt / T;
  hqT[i] = (__bf16)e[j[b * T + t] * 64 + c];
}

// fp32 tconv weight -> canonical fp32 conv weight (for the direct final conv)
__global__ void k_wtrans(const float* __restrict__ win, float* __restrict__ wout,
                         int Cout, int Cin, int K) {
  int i = blockIdx.x * blockDim.x + threadIdx.x;
  int total = Cout * Cin * K;
  if (i >= total) return;
  int k = i % K; int r = i / K; int ci = r % Cin; int co = r / Cin;
  wout[i] = win[(ci * Cout + co) * K + (K - 1 - k)];
}

// weight -> bf16 frag-packed layout wb[(co*K + k)*Cin + ci]; trans folds tconv transpose+flip
__global__ void k_wprep(const float* __restrict__ win, __bf16* __restrict__ wout,
                        int Cout, int Cin, int K, int trans) {
  int i = blockIdx.x * blockDim.x + threadIdx.x;
  int total = Cout * Cin * K;
  if (i >= total) return;
  int ci = i % Cin; int r = i / Cin; int k = r % K; int co = r / K;
  float v = trans ? win[(ci * Cout + co) * K + (K - 1 - k)]
                  : win[(co * Cin + ci) * K + k];
  wout[i] = (__bf16)v;
}

__global__ void k_finalize(const float* __restrict__ accum, float* __restrict__ out2) {
  float v = accum[0] / 65536.f;
  out2[0] = v;   // commit_loss
  out2[1] = v;   // vq_loss (numerically identical)
}

extern "C" void kernel_launch(void* const* d_in, const int* in_sizes, int n_in,
                              void* d_out, int out_size, void* d_ws, size_t ws_size,
                              hipStream_t stream) {
  const float* x     = (const float*)d_in[0];
  const float* w0    = (const float*)d_in[1];
  const float* g0_g  = (const float*)d_in[2];
  const float* g0_b  = (const float*)d_in[3];
  const float* wxa   = (const float*)d_in[4];
  const float* gxa_g = (const float*)d_in[5];
  const float* gxa_b = (const float*)d_in[6];
  const float* wxb   = (const float*)d_in[7];
  const float* gxb_g = (const float*)d_in[8];
  const float* gxb_b = (const float*)d_in[9];
  const float* wxc   = (const float*)d_in[10];
  const float* gxc_g = (const float*)d_in[11];
  const float* gxc_b = (const float*)d_in[12];
  const float* w2    = (const float*)d_in[13];
  const float* bn2_g = (const float*)d_in[14];
  const float* bn2_b = (const float*)d_in[15];
  const float* embed = (const float*)d_in[16];
  const float* wd2   = (const float*)d_in[17];
  const float* gd2_g = (const float*)d_in[18];
  const float* gd2_b = (const float*)d_in[19];
  const float* wd1a  = (const float*)d_in[20];
  const float* gd1a_g= (const float*)d_in[21];
  const float* gd1a_b= (const float*)d_in[22];
  const float* wd1b  = (const float*)d_in[23];
  const float* gd1b_g= (const float*)d_in[24];
  const float* gd1b_b= (const float*)d_in[25];
  const float* wd0   = (const float*)d_in[26];
  const float* bd0   = (const float*)d_in[27];
  float* out = (float*)d_out;

  const int B = 128, C1 = 128, C2 = 64, Dw = 64, L = 2048;
  const int BIG = B * C1 * L;  // 33,554,432 elements
  float*  A    = (float*)d_ws;
  float*  Bb   = A + BIG;
  float*  Cc   = Bb + BIG;
  float*  Dd   = Cc + BIG;
  __bf16* E    = (__bf16*)(Dd + BIG);        // bf16-T staging #1
  __bf16* F    = E + BIG;                    // bf16-T staging #2
  float*  ww   = (float*)(F + BIG);
  float*  lacc = ww + 1024;
  int*    jix  = (int*)(lacc + 32);
  float*  wt0  = (float*)(jix + 65536);      // fp32 canonical final tconv weight
  __bf16* eb   = (__bf16*)(wt0 + 4 * C1 * 7 + 8);
  __bf16* wbxa = eb + 1024 * 64;
  __bf16* wbxc = wbxa + Dw * C1;
  __bf16* wb2  = wbxc + C1 * Dw;
  __bf16* wbd2 = wb2 + C2 * C1 * 3;
  __bf16* wbd1a= wbd2 + C1 * C2 * 3;
  __bf16* wbd1b= wbd1a + C1 * C1 * 5;

  auto cdiv = [](int a, int b) { return (a + b - 1) / b; };

  // ---- weight / codebook prep ----
  { int tot = Dw * C1 * 1; k_wprep<<<cdiv(tot,256),256,0,stream>>>(wxa,  wbxa,  Dw, C1, 1, 0); }
  { int tot = C1 * Dw * 1; k_wprep<<<cdiv(tot,256),256,0,stream>>>(wxc,  wbxc,  C1, Dw, 1, 0); }
  { int tot = C2 * C1 * 3; k_wprep<<<cdiv(tot,256),256,0,stream>>>(w2,   wb2,   C2, C1, 3, 0); }
  { int tot = C1 * C2 * 3; k_wprep<<<cdiv(tot,256),256,0,stream>>>(wd2,  wbd2,  C1, C2, 3, 1); }
  { int tot = C1 * C1 * 5; k_wprep<<<cdiv(tot,256),256,0,stream>>>(wd1a, wbd1a, C1, C1, 5, 1); }
  { int tot = C1 * C1 * 5; k_wprep<<<cdiv(tot,256),256,0,stream>>>(wd1b, wbd1b, C1, C1, 5, 1); }
  { int tot = 4 * C1 * 7;  k_wtrans<<<cdiv(tot,256),256,0,stream>>>(wd0, wt0, 4, C1, 7); }
  k_eprep<<<cdiv(1024*64,256),256,0,stream>>>(embed, eb);
  k_ww<<<4,256,0,stream>>>(embed, ww);
  k_zero<<<1,32,0,stream>>>(lacc, 1);

  // ---- encoder: conv0 + GN + SELU ----
  { int tot = B * C1 * L;
    k_conv_direct<<<cdiv(tot,256),256,0,stream>>>(x, w0, nullptr, A, B, 4, C1, L, 7, 1); }
  k_gn<<<dim3(NGRP,B),256,0,stream>>>(A, nullptr, Bb, E, g0_g, g0_b, C1, L, 1);

  // ---- two ResNeXt blocks (shared weights), each followed by MaxPool(2) ----
  for (int pass = 0; pass < 2; ++pass) {
    int T = (pass == 0) ? 2048 : 1024;
    k_conv_wmma<<<dim3(T/64, Dw/16, B),32,0,stream>>>(E, wbxa, nullptr, nullptr, Cc, nullptr, C1, Dw, T, 1);
    k_gn<<<dim3(NGRP,B),256,0,stream>>>(Cc, nullptr, Dd, nullptr, gxa_g, gxa_b, Dw, T, 1);
    { int tot = B * Dw * T;
      k_conv_direct<<<cdiv(tot,256),256,0,stream>>>(Dd, wxb, nullptr, Cc, B, Dw, Dw, T, 5, 32); }
    k_gn<<<dim3(NGRP,B),256,0,stream>>>(Cc, nullptr, Dd, F, gxb_g, gxb_b, Dw, T, 1);
    k_conv_wmma<<<dim3(T/64, C1/16, B),32,0,stream>>>(F, wbxc, nullptr, nullptr, Cc, nullptr, Dw, C1, T, 1);
    k_gn<<<dim3(NGRP,B),256,0,stream>>>(Cc, Bb, A, nullptr, gxc_g, gxc_b, C1, T, 1);  // selu(h + gn(y))
    { int Tout = T / 2, tot = B * C1 * Tout;
      k_pool<<<cdiv(tot,256),256,0,stream>>>(A, Bb, E, tot, Tout, C1); }
  }

  // ---- conv2 + BatchNorm affine : (B,128,512) -> (B,64,512), bf16-T out feeds VQ ----
  k_conv_wmma<<<dim3(512/64, C2/16, B),32,0,stream>>>(E, wb2, bn2_g, bn2_b, Cc, F, C1, C2, 512, 3);

  // ---- vector quantization ----
  k_vq<<<(B*512)/16,32,0,stream>>>(F, eb, ww, 512, jix, lacc);
  { int tot = B * 64 * 512;
    k_gather<<<cdiv(tot,256),256,0,stream>>>(jix, embed, E, 512); }

  // ---- decoder: deres2 ----
  k_conv_wmma<<<dim3(512/64, C1/16, B),32,0,stream>>>(E, wbd2, nullptr, nullptr, A, nullptr, C2, C1, 512, 3);
  k_gn<<<dim3(NGRP,B),256,0,stream>>>(A, nullptr, Bb, nullptr, gd2_g, gd2_b, C1, 512, 1);

  // ---- two (upsample + deres1) stages ----
  for (int pass = 0; pass < 2; ++pass) {
    int T = (pass == 0) ? 1024 : 2048;
    { int tot = B * C1 * T;
      k_up<<<cdiv(tot,256),256,0,stream>>>(Bb, A, E, tot, T, C1); }
    k_conv_wmma<<<dim3(T/64, C1/16, B),32,0,stream>>>(E, wbd1a, nullptr, nullptr, Cc, nullptr, C1, C1, T, 5);
    k_gn<<<dim3(NGRP,B),256,0,stream>>>(Cc, nullptr, Dd, F, gd1a_g, gd1a_b, C1, T, 1);
    k_conv_wmma<<<dim3(T/64, C1/16, B),32,0,stream>>>(F, wbd1b, nullptr, nullptr, Cc, nullptr, C1, C1, T, 5);
    k_gn<<<dim3(NGRP,B),256,0,stream>>>(Cc, A, Bb, nullptr, gd1b_g, gd1b_b, C1, T, 1); // selu(h + gn(y))
  }

  // ---- final tconv (Cout=4) with bias -> recon, then losses ----
  { int tot = B * 4 * L;
    k_conv_direct<<<cdiv(tot,256),256,0,stream>>>(Bb, wt0, bd0, out, B, C1, 4, L, 7, 1); }
  k_finalize<<<1,1,0,stream>>>(lacc, out + B * 4 * L);
}
